// MultiHeadAttention_84791244358011
// MI455X (gfx1250) — matrix operand order repaired and checked by reading, not compiled, not run on hardware
//
#include <hip/hip_runtime.h>
#include <hip/hip_bf16.h>

// Problem constants (B=8, S=4096, HID=1024, NH=16, HD=64)
#define BATCH 8
#define SEQ   4096
#define NTOK  32768   // B*S
#define HID_  1024
#define NH_   16
#define HD_   64

// ---------------------------------------------------------------------------
// Types for WMMA bf16 fragments (wave32, 16x16x32, f32 accumulate)
// ---------------------------------------------------------------------------
typedef __bf16        v16bf __attribute__((ext_vector_type(16)));
typedef float         v8f   __attribute__((ext_vector_type(8)));
typedef unsigned int  v4u   __attribute__((ext_vector_type(4)));

union FragU { v16bf v; v4u q[2]; };

__device__ __forceinline__ unsigned short f2bf(float f) {
  unsigned int x = __float_as_uint(f);
  unsigned int r = x + 0x7FFFu + ((x >> 16) & 1u);   // round-to-nearest-even
  return (unsigned short)(r >> 16);
}
__device__ __forceinline__ float bf2f(unsigned short h) {
  return __uint_as_float(((unsigned int)h) << 16);
}

__device__ __forceinline__ v8f wmma_bf16(v16bf a, v16bf b, v8f c) {
  // 8 args: (neg_a, A, neg_b, B, c_mod, C, reuse_a, reuse_b)
  return __builtin_amdgcn_wmma_f32_16x16x32_bf16(false, a, false, b,
                                                 (short)0, c, false, false);
}

// ---------------------------------------------------------------------------
// CDNA5 async global->LDS copy (ASYNCcnt-tracked) + wait.
// LDS destination address taken from the lds_off VGPR; gaddr from VGPR pair (GV mode, SADDR=off).
// ---------------------------------------------------------------------------
__device__ __forceinline__ void async_copy_b128(unsigned lds_off,
                                                unsigned long long gaddr) {
  asm volatile("global_load_async_to_lds_b128 %0, %1, off"
               :: "v"(lds_off), "v"(gaddr) : "memory");
}
__device__ __forceinline__ void wait_async0() {
  asm volatile("s_wait_asynccnt 0x0" ::: "memory");
}

// ---------------------------------------------------------------------------
// fp32 -> bf16 elementwise cast (weights)
// ---------------------------------------------------------------------------
__global__ void cast_f32_bf16(const float* __restrict__ in,
                              unsigned short* __restrict__ out, int n) {
  int i = blockIdx.x * 256 + threadIdx.x;
  if (i < n) out[i] = f2bf(in[i]);
}

// ---------------------------------------------------------------------------
// LayerNorm over HID per token -> xnb (bf16), and x -> xb (bf16)
// 256 threads per token, 4 elems/thread.
// ---------------------------------------------------------------------------
__global__ void ln_cast_kernel(const float* __restrict__ x,
                               const float* __restrict__ gamma,
                               const float* __restrict__ beta,
                               unsigned short* __restrict__ xb,
                               unsigned short* __restrict__ xnb) {
  const int tok = blockIdx.x;
  const int tid = threadIdx.x;
  const float* row = x + (size_t)tok * HID_;

  float v[4];
  float s = 0.f, s2 = 0.f;
#pragma unroll
  for (int i = 0; i < 4; i++) {
    v[i] = row[tid + 256 * i];
    s  += v[i];
    s2 += v[i] * v[i];
  }
#pragma unroll
  for (int m = 16; m >= 1; m >>= 1) {
    s  += __shfl_xor(s,  m, 32);
    s2 += __shfl_xor(s2, m, 32);
  }
  __shared__ float red[16];
  const int wave = tid >> 5, lane = tid & 31;
  if (lane == 0) { red[wave] = s; red[8 + wave] = s2; }
  __syncthreads();
  if (tid < 32) {
    float a  = (tid < 8) ? red[tid] : 0.f;
    float b2 = (tid < 8) ? red[8 + tid] : 0.f;
#pragma unroll
    for (int m = 4; m >= 1; m >>= 1) {
      a  += __shfl_xor(a,  m, 32);
      b2 += __shfl_xor(b2, m, 32);
    }
    if (tid == 0) { red[0] = a; red[8] = b2; }
  }
  __syncthreads();
  const float mu   = red[0] * (1.0f / (float)HID_);
  const float var  = red[8] * (1.0f / (float)HID_) - mu * mu;
  const float rstd = rsqrtf(var + 1e-12f);

#pragma unroll
  for (int i = 0; i < 4; i++) {
    int c = tid + 256 * i;
    float xn = (v[i] - mu) * rstd * gamma[c] + beta[c];
    xnb[(size_t)tok * HID_ + c] = f2bf(xn);
    xb [(size_t)tok * HID_ + c] = f2bf(v[i]);
  }
}

// ---------------------------------------------------------------------------
// GEMM: out[M,N] = A[M,K] * W[N,K]^T + bias  (A,W bf16; f32 accumulate)
// grid (M/128, N/128), 256 threads = 8 waves (4 M-rows x 2 N-cols of waves).
// Each wave: 32x64 output = 2x4 WMMA fragments (8 accumulators).
// A tile (128x32) and W tile (128x32) double-buffered in LDS, filled with
// global_load_async_to_lds_b128 (ASYNCcnt) and consumed via ds_load_b128.
// MODE 0: store bf16 (projections). MODE 1: store f32 + bias + residual.
// ---------------------------------------------------------------------------
template <int MODE>
__global__ void gemm_bf16_wmma(const unsigned short* __restrict__ A,
                               const unsigned short* __restrict__ W,
                               const float* __restrict__ bias,
                               void* __restrict__ out,
                               const float* __restrict__ resid,
                               int M, int K, int N) {
  const int tid     = threadIdx.x;
  const int wave    = tid >> 5;
  const int lane    = tid & 31;
  const int laneRow = lane & 15;
  const int halfSel = lane >> 4;
  const int waveM   = wave >> 1;          // 0..3  -> 32-row strip
  const int waveN   = wave & 1;           // 0..1  -> 64-col strip

  const int mBlock = blockIdx.x * 128;
  const int nBlock = blockIdx.y * 128;

  // Double-buffered tiles: 128 rows x 32 k, pitch 32 elems (64B) -> 8KB each.
  __shared__ __align__(16) unsigned short sA[2][128 * 32];
  __shared__ __align__(16) unsigned short sW[2][128 * 32];
  const unsigned ldsA0 = (unsigned)(uintptr_t)&sA[0][0];
  const unsigned ldsW0 = (unsigned)(uintptr_t)&sW[0][0];

  // Stage one BK=32 slab (A + W) into buffer `buf` with async copies.
  // 512 chunks of 16B per tile; 256 threads -> 2 chunks each per tile.
  auto stage = [&](int buf, int kb) {
#pragma unroll
    for (int c = 0; c < 2; c++) {
      const int chunk = tid + c * 256;          // 0..511
      const int row   = chunk >> 2;             // 0..127
      const int cc    = chunk & 3;              // 16B chunk within 64B row
      const unsigned long long ga = (unsigned long long)(uintptr_t)
          (A + (size_t)(mBlock + row) * K + kb + cc * 8);
      const unsigned long long gw = (unsigned long long)(uintptr_t)
          (W + (size_t)(nBlock + row) * K + kb + cc * 8);
      async_copy_b128(ldsA0 + (unsigned)(buf * 8192 + row * 64 + cc * 16), ga);
      async_copy_b128(ldsW0 + (unsigned)(buf * 8192 + row * 64 + cc * 16), gw);
    }
  };

  v8f acc[2][4] = {};
  stage(0, 0);
  int cur = 0;

  for (int kb = 0; kb < K; kb += 32) {
    wait_async0();        // our async fills for buffer `cur` have landed
    __syncthreads();      // everyone's fills landed; prev reads of cur^1 done
    if (kb + 32 < K) stage(cur ^ 1, kb + 32);

    const unsigned short* aT = &sA[cur][0];
    const unsigned short* wT = &sW[cur][0];

    FragU af[2];
#pragma unroll
    for (int mf = 0; mf < 2; mf++) {
      const unsigned short* ar = aT + (waveM * 32 + mf * 16 + laneRow) * 32;
      af[mf].q[0] = *(const v4u*)(ar + 8 * halfSel);
      af[mf].q[1] = *(const v4u*)(ar + 16 + 8 * halfSel);
    }
#pragma unroll
    for (int nf = 0; nf < 4; nf++) {
      const unsigned short* br =
          wT + (waveN * 64 + nf * 16 + laneRow) * 32;
      FragU bfr;
      bfr.q[0] = *(const v4u*)(br + 8 * halfSel);
      bfr.q[1] = *(const v4u*)(br + 16 + 8 * halfSel);
#pragma unroll
      for (int mf = 0; mf < 2; mf++)
        acc[mf][nf] = wmma_bf16(af[mf].v, bfr.v, acc[mf][nf]);
    }
    cur ^= 1;
  }

#pragma unroll
  for (int mf = 0; mf < 2; mf++) {
#pragma unroll
    for (int nf = 0; nf < 4; nf++) {
      const int col  = nBlock + waveN * 64 + nf * 16 + laneRow;
      const float bc = bias[col];
#pragma unroll
      for (int r = 0; r < 8; r++) {
        const size_t row =
            (size_t)(mBlock + waveM * 32 + mf * 16 + r + 8 * halfSel);
        const float val = acc[mf][nf][r] + bc;
        if (MODE == 0) {
          ((unsigned short*)out)[row * N + col] = f2bf(val);
        } else {
          ((float*)out)[row * N + col] = val + resid[row * N + col];
        }
      }
    }
  }
}

// ---------------------------------------------------------------------------
// ELU + per-(token,head) L2 normalize, in place on bf16 q/k.
// One wave per (token,head); 64 values -> 2 per lane. extraScale folds 1/sqrt(HD).
// ---------------------------------------------------------------------------
__global__ void elu_norm_kernel(unsigned short* __restrict__ qk, float extraScale) {
  const int gwave = (blockIdx.x * blockDim.x + threadIdx.x) >> 5;
  const int lane  = threadIdx.x & 31;
  const int tok   = gwave >> 4;   // / NH
  const int head  = gwave & 15;

  unsigned short* p = qk + (size_t)tok * HID_ + head * HD_;
  float v0 = bf2f(p[lane]);
  float v1 = bf2f(p[lane + 32]);
  v0 = v0 > 0.f ? v0 : expm1f(v0);
  v1 = v1 > 0.f ? v1 : expm1f(v1);
  float ss = v0 * v0 + v1 * v1;
#pragma unroll
  for (int m = 16; m >= 1; m >>= 1) ss += __shfl_xor(ss, m, 32);
  const float sc = rsqrtf(ss) * extraScale;
  p[lane]      = f2bf(v0 * sc);
  p[lane + 32] = f2bf(v1 * sc);
}

// ---------------------------------------------------------------------------
// kv[b,h,d,e] = sum_s ek[b,s,h,d] * v[b,s,h,e]  (64x64 out, K=4096)
// One block per (b,h), 128 threads = 4 waves; ek/v tiles (64s x 64d) staged
// TRANSPOSED into LDS so fragment reads are contiguous. pitch 80 keeps 16B align.
// ---------------------------------------------------------------------------
__global__ void kv_kernel(const unsigned short* __restrict__ ek,
                          const unsigned short* __restrict__ vv,
                          unsigned short* __restrict__ kv) {
  const int bh = blockIdx.x;            // 0..127
  const int b  = bh >> 4;
  const int h  = bh & 15;
  const int tid     = threadIdx.x;      // 128
  const int wave    = tid >> 5;
  const int lane    = tid & 31;
  const int laneRow = lane & 15;
  const int halfSel = lane >> 4;

  __shared__ __align__(16) unsigned short ekT[64 * 80];
  __shared__ __align__(16) unsigned short vT [64 * 80];

  v8f acc[4] = {};
  const size_t baseTok = (size_t)b * SEQ;

  for (int s0 = 0; s0 < SEQ; s0 += 64) {
    __syncthreads();
    for (int i = tid; i < 64 * 64; i += 128) {
      const int s = i >> 6;           // global-coalesced read over d
      const int d = i & 63;
      const size_t g = (baseTok + s0 + s) * HID_ + h * HD_ + d;
      ekT[d * 80 + s] = ek[g];
      vT [d * 80 + s] = vv[g];
    }
    __syncthreads();
#pragma unroll
    for (int kb = 0; kb < 64; kb += 32) {
      FragU af;
      const unsigned short* ar = &ekT[(wave * 16 + laneRow) * 80 + kb];
      af.q[0] = *(const v4u*)(ar + 8 * halfSel);
      af.q[1] = *(const v4u*)(ar + 16 + 8 * halfSel);
#pragma unroll
      for (int nf = 0; nf < 4; nf++) {
        const unsigned short* br = &vT[(nf * 16 + laneRow) * 80 + kb];
        FragU bfr;
        bfr.q[0] = *(const v4u*)(br + 8 * halfSel);
        bfr.q[1] = *(const v4u*)(br + 16 + 8 * halfSel);
        acc[nf] = wmma_bf16(af.v, bfr.v, acc[nf]);
      }
    }
  }

#pragma unroll
  for (int nf = 0; nf < 4; nf++) {
    const int col = nf * 16 + laneRow;
#pragma unroll
    for (int r = 0; r < 8; r++) {
      const int row = wave * 16 + r + 8 * halfSel;
      kv[((size_t)bh * 64 + row) * 64 + col] = f2bf(acc[nf][r]);
    }
  }
}

// ---------------------------------------------------------------------------
// ctx[t, h*64+e] = sum_d eq[t, h*64+d] * kv[b,h,d,e]  (1/sqrt(HD) folded in eq)
// grid (B*NH, SEQ/128); 256 threads = 8 waves; kv staged transposed in LDS.
// ---------------------------------------------------------------------------
__global__ void ctx_kernel(const unsigned short* __restrict__ eq,
                           const unsigned short* __restrict__ kv,
                           unsigned short* __restrict__ ctx) {
  const int bh = blockIdx.x;
  const int b  = bh >> 4, h = bh & 15;
  const int tid     = threadIdx.x;      // 256
  const int wave    = tid >> 5;
  const int lane    = tid & 31;
  const int laneRow = lane & 15;
  const int halfSel = lane >> 4;

  __shared__ __align__(16) unsigned short kvT[64 * 80];
  for (int i = tid; i < 64 * 64; i += 256) {
    const int d = i >> 6, e = i & 63;
    kvT[e * 80 + d] = kv[(size_t)bh * 4096 + d * 64 + e];
  }
  __syncthreads();

  const size_t tokBase = (size_t)b * SEQ + blockIdx.y * 128 + wave * 16;
  const unsigned short* aRow = eq + (tokBase + laneRow) * HID_ + h * HD_;

  v8f acc[4] = {};
#pragma unroll
  for (int kb = 0; kb < 64; kb += 32) {
    FragU af;
    af.q[0] = *(const v4u*)(aRow + kb + 8 * halfSel);
    af.q[1] = *(const v4u*)(aRow + kb + 16 + 8 * halfSel);
#pragma unroll
    for (int nf = 0; nf < 4; nf++) {
      const unsigned short* br = &kvT[(nf * 16 + laneRow) * 80 + kb];
      FragU bfr;
      bfr.q[0] = *(const v4u*)(br + 8 * halfSel);
      bfr.q[1] = *(const v4u*)(br + 16 + 8 * halfSel);
      acc[nf] = wmma_bf16(af.v, bfr.v, acc[nf]);
    }
  }

#pragma unroll
  for (int nf = 0; nf < 4; nf++) {
    const int col = nf * 16 + laneRow;
#pragma unroll
    for (int r = 0; r < 8; r++) {
      const size_t row = tokBase + r + 8 * halfSel;
      ctx[row * HID_ + h * HD_ + col] = f2bf(acc[nf][r]);
    }
  }
}

// ---------------------------------------------------------------------------
// Launch. Workspace layout (all 256B aligned), total ~329 MiB:
//   xb, xnb(->reused as ctx), qb, kb, vb : 64 MiB each (bf16 [NTOK,HID])
//   wqb,wkb,wvb,wob : 2 MiB each ; kvb : 1 MiB
// ---------------------------------------------------------------------------
extern "C" void kernel_launch(void* const* d_in, const int* in_sizes, int n_in,
                              void* d_out, int out_size, void* d_ws, size_t ws_size,
                              hipStream_t stream) {
  const float* x     = (const float*)d_in[0];
  // d_in[1] = attention_mask (unused, as in the reference)
  const float* gamma = (const float*)d_in[2];
  const float* beta  = (const float*)d_in[3];
  const float* Wq    = (const float*)d_in[4];
  const float* bq    = (const float*)d_in[5];
  const float* Wk    = (const float*)d_in[6];
  const float* bk    = (const float*)d_in[7];
  const float* Wv    = (const float*)d_in[8];
  const float* bv    = (const float*)d_in[9];
  const float* Wo    = (const float*)d_in[10];
  const float* bo    = (const float*)d_in[11];

  char* ws = (char*)d_ws;
  size_t off = 0;
  auto alloc = [&](size_t bytes) -> void* {
    void* p = ws + off;
    off += (bytes + 255) & ~(size_t)255;
    return p;
  };
  const size_t TOKB = (size_t)NTOK * HID_ * sizeof(unsigned short);
  unsigned short* xb  = (unsigned short*)alloc(TOKB);
  unsigned short* xnb = (unsigned short*)alloc(TOKB);   // later reused as ctx
  unsigned short* qb  = (unsigned short*)alloc(TOKB);
  unsigned short* kbf = (unsigned short*)alloc(TOKB);
  unsigned short* vb  = (unsigned short*)alloc(TOKB);
  unsigned short* wqb = (unsigned short*)alloc((size_t)HID_ * HID_ * 2);
  unsigned short* wkb = (unsigned short*)alloc((size_t)HID_ * HID_ * 2);
  unsigned short* wvb = (unsigned short*)alloc((size_t)HID_ * HID_ * 2);
  unsigned short* wob = (unsigned short*)alloc((size_t)HID_ * HID_ * 2);
  unsigned short* kvb = (unsigned short*)alloc((size_t)BATCH * NH_ * HD_ * HD_ * 2);

  const int wn = HID_ * HID_;
  cast_f32_bf16<<<wn / 256, 256, 0, stream>>>(Wq, wqb, wn);
  cast_f32_bf16<<<wn / 256, 256, 0, stream>>>(Wk, wkb, wn);
  cast_f32_bf16<<<wn / 256, 256, 0, stream>>>(Wv, wvb, wn);
  cast_f32_bf16<<<wn / 256, 256, 0, stream>>>(Wo, wob, wn);

  ln_cast_kernel<<<NTOK, 256, 0, stream>>>(x, gamma, beta, xb, xnb);

  dim3 gg(NTOK / 128, HID_ / 128);
  gemm_bf16_wmma<0><<<gg, 256, 0, stream>>>(xnb, wqb, bq, qb,  nullptr, NTOK, HID_, HID_);
  gemm_bf16_wmma<0><<<gg, 256, 0, stream>>>(xb,  wkb, bk, kbf, nullptr, NTOK, HID_, HID_);
  gemm_bf16_wmma<0><<<gg, 256, 0, stream>>>(xb,  wvb, bv, vb,  nullptr, NTOK, HID_, HID_);

  // 8 waves/block; one wave per (token, head). Fold 1/sqrt(HD)=0.125 into eq.
  elu_norm_kernel<<<(NTOK * NH_) / 8, 256, 0, stream>>>(qb,  0.125f);
  elu_norm_kernel<<<(NTOK * NH_) / 8, 256, 0, stream>>>(kbf, 1.0f);

  kv_kernel<<<BATCH * NH_, 128, 0, stream>>>(kbf, vb, kvb);

  dim3 cg(BATCH * NH_, SEQ / 128);
  ctx_kernel<<<cg, 256, 0, stream>>>(qb, kvb, xnb /* ctx reuses xnb */);

  gemm_bf16_wmma<1><<<gg, 256, 0, stream>>>(xnb, wob, bo, d_out, x, NTOK, HID_, HID_);
}
